// Mamba2Simple_17557826306471
// MI455X (gfx1250) — compile-verified, matrix operand order checked by reading, not run on hardware
//
#include <hip/hip_runtime.h>
#include <hip/hip_bf16.h>

// ---- model constants (match reference) ----
#define BATCH      2
#define SEQLEN     2048
#define BLr        4096        // BATCH*SEQLEN
#define IN_CH      256
#define OUT_CH     256
#define DMODEL     1024
#define DSTATE     128
#define HEADDIM    64
#define DINNER     2048
#define NHEADS     32
#define CONVDIM    2304        // DINNER + 2*DSTATE
#define DINPROJ    4384        // 2*DINNER + 2*DSTATE + NHEADS
#define DINPROJ_P  4480        // padded to multiple of 128
#define NBLOCKS    2
#define EPSV       1e-5f

// ---- WMMA vector types (CDNA5, wave32) ----
typedef __attribute__((ext_vector_type(16))) __bf16 v16bf;
typedef __attribute__((ext_vector_type(8)))  __bf16 v8bf;
typedef __attribute__((ext_vector_type(4)))  __bf16 v4bf;
typedef __attribute__((ext_vector_type(8)))  float  v8f;
typedef __attribute__((ext_vector_type(4)))  float  v4f;

// CDNA5 async global->LDS copy (128b per lane), tracked by ASYNCcnt.
// lds_off: byte offset within wave's LDS (low 32 bits of flat shared addr).
#define ASYNC_B128(lds_off, gptr)                                        \
  asm volatile("global_load_async_to_lds_b128 %0, %1, off"               \
               :: "v"(lds_off), "v"(gptr) : "memory")
#define WAIT_ASYNC0() asm volatile("s_wait_asynccnt 0x0" ::: "memory")

// =================== fp32 -> bf16 streaming convert (vectorized) =============
__global__ __launch_bounds__(256)
void convert_bf16(const float* __restrict__ src, __bf16* __restrict__ dst, int n4) {
  const int stride = gridDim.x * blockDim.x;
  for (int i = blockIdx.x * blockDim.x + threadIdx.x; i < n4; i += stride) {
    v4f v = ((const v4f*)src)[i];
    v4bf o;
    o[0] = (__bf16)v[0]; o[1] = (__bf16)v[1];
    o[2] = (__bf16)v[2]; o[3] = (__bf16)v[3];
    ((v4bf*)dst)[i] = o;
  }
}

// ====== weight convert + transpose + pad: B[K,Nreal] f32 -> Bt[Npad,K] bf16 ==
__global__ __launch_bounds__(256)
void convert_transpose_bf16(const float* __restrict__ B, __bf16* __restrict__ Bt,
                            int K, int Nreal) {
  __shared__ __bf16 tile[32][33];
  const int k0 = blockIdx.x * 32;
  const int n0 = blockIdx.y * 32;
  const int tx = threadIdx.x & 31;
  const int ty = threadIdx.x >> 5;
  #pragma unroll
  for (int i = 0; i < 4; ++i) {
    int k = k0 + ty + i * 8;
    int n = n0 + tx;
    float v = (n < Nreal) ? B[(size_t)k * Nreal + n] : 0.0f;
    tile[ty + i * 8][tx] = (__bf16)v;
  }
  __syncthreads();
  #pragma unroll
  for (int i = 0; i < 4; ++i) {
    int n = n0 + ty + i * 8;      // row of Bt
    int k = k0 + tx;
    Bt[(size_t)n * K + k] = tile[tx][ty + i * 8];
  }
}

// ============ bf16 WMMA GEMM: C[M,Nreal] = A[M,K] * Bt[Npad,K]^T (+bias) =====
// Double-buffered LDS, tiles staged by global_load_async_to_lds_b128 (no VGPR
// staging -> no spill). K multiple of 64, M multiple of 128, Npad=gridDim.x*128.
#define BM 128
#define BN 128
#define BK 64
#define BUFBYTES (BM * BK * 2)   // 16384 bytes per buffer (A and B alike)

__global__ __launch_bounds__(256)
void wmma_gemm_bf16(const __bf16* __restrict__ A, const __bf16* __restrict__ Bt,
                    const float* __restrict__ bias, float* __restrict__ C,
                    int M, int Nreal, int K) {
  __shared__ __bf16 As[2][BM][BK];   // 16 KB x2
  __shared__ __bf16 Bs[2][BN][BK];   // 16 KB x2  (Bs[n][k])

  const int tid  = threadIdx.x;
  const int wave = tid >> 5;
  const int lane = tid & 31;
  const int wm = wave >> 1;          // 0..3 : rows [wm*32, wm*32+32)
  const int wn = wave & 1;           // 0..1 : cols [wn*64, wn*64+64)
  const int block_m = blockIdx.y * BM;
  const int block_n = blockIdx.x * BN;
  const int m_frag = lane & 15;
  const int k_half = (lane >> 4) * 8;

  // staging coordinates: 128x64 bf16 tile = 1024 8-elem vectors, 4 per thread
  const __bf16* aP[4];
  const __bf16* bP[4];
  unsigned aL[4], bL[4];
  #pragma unroll
  for (int j = 0; j < 4; ++j) {
    int vi = tid + j * 256;
    int r = vi >> 3;
    int c = (vi & 7) * 8;
    aP[j] = A  + (size_t)(block_m + r) * K + c;
    bP[j] = Bt + (size_t)(block_n + r) * K + c;
    aL[j] = (unsigned)(size_t)&As[0][r][c];
    bL[j] = (unsigned)(size_t)&Bs[0][r][c];
  }

  v8f acc[8];
  #pragma unroll
  for (int t = 0; t < 8; ++t)
    #pragma unroll
    for (int j = 0; j < 8; ++j) acc[t][j] = 0.0f;

  // ---- prologue: async-stage tile 0 into buffer 0 ----
  #pragma unroll
  for (int j = 0; j < 4; ++j) {
    ASYNC_B128(aL[j], aP[j]);
    ASYNC_B128(bL[j], bP[j]);
    aP[j] += BK;
    bP[j] += BK;
  }
  WAIT_ASYNC0();
  __syncthreads();

  const int ntiles = K / BK;
  for (int kt = 0; kt < ntiles; ++kt) {
    const int cur = kt & 1;
    // issue next tile's async copies into the other buffer (overlaps WMMA)
    if (kt + 1 < ntiles) {
      const unsigned boff = (unsigned)((kt + 1) & 1) * (unsigned)BUFBYTES;
      #pragma unroll
      for (int j = 0; j < 4; ++j) {
        ASYNC_B128(aL[j] + boff, aP[j]);
        ASYNC_B128(bL[j] + boff, bP[j]);
        aP[j] += BK;
        bP[j] += BK;
      }
    }
    // ---- compute on current buffer: 2 k-steps of 32, 8 wmma each ----
    #pragma unroll
    for (int kk = 0; kk < BK; kk += 32) {
      v16bf af[2];
      #pragma unroll
      for (int tm = 0; tm < 2; ++tm) {
        const __bf16* ap = &As[cur][wm * 32 + tm * 16 + m_frag][kk];
        v8bf lo = *(const v8bf*)(ap + k_half);
        v8bf hi = *(const v8bf*)(ap + 16 + k_half);
        af[tm] = __builtin_shufflevector(lo, hi,
            0,1,2,3,4,5,6,7,8,9,10,11,12,13,14,15);
      }
      #pragma unroll
      for (int tn = 0; tn < 4; ++tn) {
        const __bf16* bp = &Bs[cur][wn * 64 + tn * 16 + m_frag][kk];
        v8bf lo = *(const v8bf*)(bp + k_half);
        v8bf hi = *(const v8bf*)(bp + 16 + k_half);
        v16bf bfrag = __builtin_shufflevector(lo, hi,
            0,1,2,3,4,5,6,7,8,9,10,11,12,13,14,15);
        acc[0 * 4 + tn] = __builtin_amdgcn_wmma_f32_16x16x32_bf16(
            false, af[0], false, bfrag, (short)0, acc[0 * 4 + tn], false, false);
        acc[1 * 4 + tn] = __builtin_amdgcn_wmma_f32_16x16x32_bf16(
            false, af[1], false, bfrag, (short)0, acc[1 * 4 + tn], false, false);
      }
    }
    // own async copies must land before anyone reads the other buffer
    WAIT_ASYNC0();
    __syncthreads();
  }

  // ---- store C: VGPR r -> M = r + (lane/16)*8 ; N = lane%16 ----
  #pragma unroll
  for (int tm = 0; tm < 2; ++tm)
    #pragma unroll
    for (int tn = 0; tn < 4; ++tn) {
      int gn = block_n + wn * 64 + tn * 16 + (lane & 15);
      if (gn >= Nreal) continue;
      float bv = bias ? bias[gn] : 0.0f;
      int gm0 = block_m + wm * 32 + tm * 16 + (lane >> 4) * 8;
      #pragma unroll
      for (int r = 0; r < 8; ++r)
        C[(size_t)(gm0 + r) * Nreal + gn] = acc[tm * 4 + tn][r] + bv;
    }
}

// ============ depthwise causal conv (K=4) + SiLU on xBC, softplus(dt) ========
__global__ __launch_bounds__(256)
void conv_silu_dt(const float* __restrict__ zx, const float* __restrict__ conv_w,
                  const float* __restrict__ conv_b, const float* __restrict__ dt_bias,
                  float* __restrict__ xbc_out, float* __restrict__ dt_out) {
  const int stride = gridDim.x * blockDim.x;
  const int totalX = BLr * CONVDIM;
  for (int idx = blockIdx.x * blockDim.x + threadIdx.x; idx < totalX; idx += stride) {
    int c  = idx % CONVDIM;
    int bl = idx / CONVDIM;
    int l  = bl % SEQLEN;
    float acc = conv_b[c];
    #pragma unroll
    for (int k = 0; k < 4; ++k) {
      int lk = l + k - 3;
      if (lk >= 0)
        acc += zx[(size_t)(bl + k - 3) * DINPROJ + DINNER + c] * conv_w[c * 4 + k];
    }
    float s = 1.0f / (1.0f + __expf(-acc));
    xbc_out[idx] = acc * s;
  }
  const int totalD = BLr * NHEADS;
  for (int idx = blockIdx.x * blockDim.x + threadIdx.x; idx < totalD; idx += stride) {
    int h  = idx % NHEADS;
    int bl = idx / NHEADS;
    float v = zx[(size_t)bl * DINPROJ + (2 * DINNER + 2 * DSTATE) + h] + dt_bias[h];
    dt_out[idx] = (v > 20.0f) ? v : log1pf(__expf(v));
  }
}

// ====================== selective scan (sequential in L) =====================
__global__ __launch_bounds__(256)
void ssm_scan(const float* __restrict__ xbc, const float* __restrict__ dt,
              const float* __restrict__ A_log, const float* __restrict__ Dvec,
              float* __restrict__ y) {
  const int b = blockIdx.x / NHEADS;
  const int h = blockIdx.x % NHEADS;
  const float A  = -__expf(A_log[h]);
  const float Dh = Dvec[h];
  const int tid = threadIdx.x;
  const int p = tid >> 2;
  const int nbase = (tid & 3) * 32;

  float s[32];
  #pragma unroll
  for (int i = 0; i < 32; ++i) s[i] = 0.0f;

  __shared__ float sB[DSTATE], sC[DSTATE], sX[HEADDIM], ypart[HEADDIM * 4];

  for (int l = 0; l < SEQLEN; ++l) {
    const float* row = xbc + (size_t)(b * SEQLEN + l) * CONVDIM;
    if (tid < DSTATE)  sB[tid]          = row[DINNER + tid];
    else               sC[tid - DSTATE] = row[DINNER + DSTATE + (tid - DSTATE)];
    if (tid < HEADDIM) sX[tid]          = row[h * HEADDIM + tid];
    __syncthreads();

    const float dtv  = dt[(size_t)(b * SEQLEN + l) * NHEADS + h];
    const float dA   = __expf(dtv * A);
    const float coef = dtv * sX[p];
    float acc = 0.0f;
    #pragma unroll
    for (int i = 0; i < 32; ++i) {
      float sn = s[i] * dA + coef * sB[nbase + i];
      s[i] = sn;
      acc += sn * sC[nbase + i];
    }
    ypart[p * 4 + (tid & 3)] = acc;
    __syncthreads();

    if (tid < HEADDIM) {
      float yv = ypart[tid * 4] + ypart[tid * 4 + 1] +
                 ypart[tid * 4 + 2] + ypart[tid * 4 + 3] + Dh * sX[tid];
      y[((size_t)(b * SEQLEN + l) * NHEADS + h) * HEADDIM + tid] = yv;
    }
    __syncthreads();
  }
}

// =================== gated RMSNorm: y = norm(y*silu(z))*w ====================
__global__ __launch_bounds__(256)
void gate_rmsnorm(const float* __restrict__ zx, const float* __restrict__ rms_w,
                  float* __restrict__ y) {
  const size_t bl = blockIdx.x;
  const float* zrow = zx + bl * (size_t)DINPROJ;
  float* yrow = y + bl * (size_t)DINNER;
  __shared__ float red[256];

  float vals[8];
  float ss = 0.0f;
  #pragma unroll
  for (int i = 0; i < 8; ++i) {
    int c = threadIdx.x + i * 256;
    float z = zrow[c];
    float g = z / (1.0f + __expf(-z));
    float v = yrow[c] * g;
    vals[i] = v;
    ss += v * v;
  }
  red[threadIdx.x] = ss;
  __syncthreads();
  for (int s = 128; s > 0; s >>= 1) {
    if ((int)threadIdx.x < s) red[threadIdx.x] += red[threadIdx.x + s];
    __syncthreads();
  }
  const float scale = rsqrtf(red[0] / (float)DINNER + EPSV);
  #pragma unroll
  for (int i = 0; i < 8; ++i) {
    int c = threadIdx.x + i * 256;
    yrow[c] = vals[i] * scale * rms_w[c];
  }
}

// ================================ launch =====================================
static inline void launch_gemm(const __bf16* A, const __bf16* Bt, const float* bias,
                               float* C, int M, int Nreal, int K, hipStream_t s) {
  int Npad = (Nreal + 127) & ~127;
  dim3 g(Npad / BN, M / BM);
  wmma_gemm_bf16<<<g, 256, 0, s>>>(A, Bt, bias, C, M, Nreal, K);
}

static inline void launch_convert(const float* src, __bf16* dst, size_t n,
                                  hipStream_t s) {
  int n4 = (int)(n / 4);
  int blocks = (n4 + 255) / 256;
  if (blocks > 4096) blocks = 4096;
  convert_bf16<<<blocks, 256, 0, s>>>(src, dst, n4);
}

static inline void launch_transpose(const float* B, __bf16* Bt, int K, int Nreal,
                                    hipStream_t s) {
  int Npad = (Nreal + 127) & ~127;
  dim3 g(K / 32, Npad / 32);
  convert_transpose_bf16<<<g, 256, 0, s>>>(B, Bt, K, Nreal);
}

extern "C" void kernel_launch(void* const* d_in, const int* in_sizes, int n_in,
                              void* d_out, int out_size, void* d_ws, size_t ws_size,
                              hipStream_t stream) {
  (void)in_sizes; (void)n_in; (void)out_size; (void)ws_size;
  const float* x        = (const float*)d_in[0];
  const float* Win_o    = (const float*)d_in[1];
  const float* bin_o    = (const float*)d_in[2];
  const float* Wout_o   = (const float*)d_in[3];
  const float* bout_o   = (const float*)d_in[4];
  const float* blk_Win  = (const float*)d_in[5];   // [2,1024,4384]
  const float* blk_cw   = (const float*)d_in[6];   // [2,2304,4]
  const float* blk_cb   = (const float*)d_in[7];   // [2,2304]
  const float* blk_dtb  = (const float*)d_in[8];   // [2,32]
  const float* blk_Alog = (const float*)d_in[9];   // [2,32]
  const float* blk_D    = (const float*)d_in[10];  // [2,32]
  const float* blk_rmsw = (const float*)d_in[11];  // [2,2048]
  const float* blk_Wout = (const float*)d_in[12];  // [2,2048,1024]
  float* out = (float*)d_out;

  // ---- workspace carve: fp32 region ----
  float* ws = (float*)d_ws;
  float* h0   = ws;  ws += (size_t)BLr * DMODEL;
  float* h1   = ws;  ws += (size_t)BLr * DMODEL;
  float* zx   = ws;  ws += (size_t)BLr * DINPROJ;
  float* xbc  = ws;  ws += (size_t)BLr * CONVDIM;
  float* dtb  = ws;  ws += (size_t)BLr * NHEADS;
  float* ybuf = ws;  ws += (size_t)BLr * DINNER;

  // ---- bf16 region (16B aligned: fp32 region byte count is multiple of 16) --
  __bf16* bws = (__bf16*)ws;
  __bf16* xbf    = bws;  bws += (size_t)BLr * IN_CH;
  __bf16* hbf    = bws;  bws += (size_t)BLr * DMODEL;
  __bf16* ybf    = bws;  bws += (size_t)BLr * DINNER;
  __bf16* WtIn   = bws;  bws += (size_t)DMODEL * IN_CH;          // [1024][256]
  __bf16* WtBlkI = bws;  bws += (size_t)NBLOCKS * DINPROJ_P * DMODEL;
  __bf16* WtBlkO = bws;  bws += (size_t)NBLOCKS * DMODEL * DINNER;
  __bf16* WtOut  = bws;  bws += (size_t)OUT_CH * DMODEL;         // [256][1024]

  // ---- weight conversion (transpose + pad, bf16) ----
  launch_transpose(Win_o, WtIn, IN_CH, DMODEL, stream);
  for (int b = 0; b < NBLOCKS; ++b) {
    launch_transpose(blk_Win + (size_t)b * DMODEL * DINPROJ,
                     WtBlkI + (size_t)b * DINPROJ_P * DMODEL, DMODEL, DINPROJ, stream);
    launch_transpose(blk_Wout + (size_t)b * DINNER * DMODEL,
                     WtBlkO + (size_t)b * DMODEL * DINNER, DINNER, DMODEL, stream);
  }
  launch_transpose(Wout_o, WtOut, DMODEL, OUT_CH, stream);

  // ---- 1) h = x @ Win_o + bin_o ----
  launch_convert(x, xbf, (size_t)BLr * IN_CH, stream);
  launch_gemm(xbf, WtIn, bin_o, h0, BLr, DMODEL, IN_CH, stream);

  float* hcur = h0;
  float* hnxt = h1;
  for (int blk = 0; blk < NBLOCKS; ++blk) {
    // 2a) zxbcdt = h @ Win[blk]
    launch_convert(hcur, hbf, (size_t)BLr * DMODEL, stream);
    launch_gemm(hbf, WtBlkI + (size_t)blk * DINPROJ_P * DMODEL, nullptr, zx,
                BLr, DINPROJ, DMODEL, stream);
    // 2b) causal conv + silu, softplus(dt)
    conv_silu_dt<<<(BLr * CONVDIM + 255) / 256, 256, 0, stream>>>(
        zx, blk_cw + (size_t)blk * CONVDIM * 4, blk_cb + (size_t)blk * CONVDIM,
        blk_dtb + (size_t)blk * NHEADS, xbc, dtb);
    // 2c) selective scan
    ssm_scan<<<BATCH * NHEADS, 256, 0, stream>>>(
        xbc, dtb, blk_Alog + (size_t)blk * NHEADS, blk_D + (size_t)blk * NHEADS, ybuf);
    // 2d) gated RMSNorm (in-place)
    gate_rmsnorm<<<BLr, 256, 0, stream>>>(zx, blk_rmsw + (size_t)blk * DINNER, ybuf);
    // 2e) h_next = y @ Wout[blk]
    launch_convert(ybuf, ybf, (size_t)BLr * DINNER, stream);
    launch_gemm(ybf, WtBlkO + (size_t)blk * DMODEL * DINNER, nullptr, hnxt,
                BLr, DMODEL, DINNER, stream);
    float* t = hcur; hcur = hnxt; hnxt = t;
  }

  // ---- 3) out = h @ Wout_o + bout_o ----
  launch_convert(hcur, hbf, (size_t)BLr * DMODEL, stream);
  launch_gemm(hbf, WtOut, bout_o, out, BLr, OUT_CH, DMODEL, stream);
}
